// CMP_67181878444960
// MI455X (gfx1250) — matrix operand order repaired and checked by reading, not compiled, not run on hardware
//
#include <hip/hip_runtime.h>
#include <hip/hip_bf16.h>
#include <stdint.h>

// ---------------- CDNA5 WMMA types ----------------
typedef __attribute__((ext_vector_type(16))) __bf16          v16bf;
typedef __attribute__((ext_vector_type(8)))  float           v8f;
typedef __attribute__((ext_vector_type(8)))  unsigned short  v8us;
typedef __attribute__((ext_vector_type(4)))  int             v4i;

union BF16Frag {
    v8us  h[2];
    v16bf v;
};

// Problem constants
#define NV    1024      // images (graph vertices)
#define NC    16        // base channels
#define HW    1024      // 32*32 spatial positions
#define CIN   48        // concat channels
#define CPAD  64        // channel padding for K-chunks of 32

// Async global->LDS staging (gfx1250 GLOBAL_LOAD_ASYNC_TO_LDS_B128, ASYNCcnt)
#if defined(__gfx1250__) && __has_builtin(__builtin_amdgcn_global_load_async_to_lds_b128) && \
    __has_builtin(__builtin_amdgcn_s_wait_asynccnt)
#define USE_ASYNC_LDS 1
typedef __attribute__((address_space(1))) v4i gv4i;   // global int4 (prints as __device__)
typedef __attribute__((address_space(3))) v4i lv4i;   // LDS int4    (prints as __shared__)
#else
#define USE_ASYNC_LDS 0
#endif

__device__ __forceinline__ unsigned short f2bf(float f) {
    __hip_bfloat16 h = __float2bfloat16(f);        // hardware RNE cvt
    return *(unsigned short*)&h;
}
__device__ __forceinline__ float bf2f(unsigned short u) {
    return __uint_as_float(((unsigned)u) << 16);
}

// ---------------- 1. signed scatter pooling (fp32 atomics) ----------------
__global__ __launch_bounds__(256) void pool_edges(
    const float* __restrict__ feats, const int* __restrict__ edges,
    float* __restrict__ ppos, float* __restrict__ pneg)
{
    const int e = blockIdx.x;                       // 0..4095
    const int i = blockIdx.y * 256 + threadIdx.x;   // 0..16383 (C*H*W)
    const int s = edges[e * 3 + 0];
    const int l = edges[e * 3 + 1];
    const int d = edges[e * 3 + 2];
    if (l == 0) return;
    float* dstb = (l > 0) ? ppos : pneg;
    const float fs = feats[(size_t)s * (NC * HW) + i];
    const float fd = feats[(size_t)d * (NC * HW) + i];
    atomicAdd(&dstb[(size_t)d * (NC * HW) + i], fs);
    atomicAdd(&dstb[(size_t)s * (NC * HW) + i], fd);
}

// ---------------- 2. pack x = concat(feats, pos, neg) -> channel-minor bf16 ----------------
__global__ __launch_bounds__(256) void pack_x(
    const float* __restrict__ feats, const float* __restrict__ ppos,
    const float* __restrict__ pneg, unsigned short* __restrict__ x0)
{
    size_t idx = (size_t)blockIdx.x * 256 + threadIdx.x;
    const size_t total = (size_t)NV * HW * CIN;
    if (idx >= total) return;
    const int cc = (int)(idx % CIN);
    const size_t t = idx / CIN;
    const int p = (int)(t % HW);
    const int v = (int)(t / HW);
    float val;
    if (cc < 16)      val = feats[((size_t)v * NC + cc) * HW + p];
    else if (cc < 32) val = ppos[((size_t)v * NC + (cc - 16)) * HW + p];
    else              val = pneg[((size_t)v * NC + (cc - 32)) * HW + p];
    x0[((size_t)v * HW + p) * CPAD + cc] = f2bf(val);
}

// ---------------- 3. pack weights (O,I,3,3) fp32 -> [tap][co][ci64] bf16 ----------------
__global__ __launch_bounds__(256) void pack_w(
    const float* __restrict__ w, unsigned short* __restrict__ wpk, int OC, int IC)
{
    const int idx = blockIdx.x * 256 + threadIdx.x;
    const int total = 9 * OC * CPAD;
    if (idx >= total) return;
    const int ci  = idx % CPAD;
    const int t   = idx / CPAD;
    const int co  = t % OC;
    const int tap = t / OC;     // tap = ky*3+kx
    float val = 0.f;
    if (ci < IC) val = w[((size_t)co * IC + ci) * 9 + tap];
    wpk[idx] = f2bf(val);
}

// ---------------- 4. 3x3 conv via bf16 WMMA (implicit GEMM over 9 taps) ----------------
// MODE 0: out = relu(conv+b)   -> bf16 channel-minor
// MODE 1: out = resid + conv+b -> bf16 channel-minor
// MODE 2: out = conv+b         -> fp32 NCHW (final conv, OC=16)
template<int MODE, int NTILES, int OC>
__global__ __launch_bounds__(128) void conv3x3_wmma(
    const unsigned short* __restrict__ xin,   // [V][1024][CPAD] bf16
    const unsigned short* __restrict__ wpk,   // [9][OC][CPAD]  bf16
    const float* __restrict__ bias,           // [OC]
    const unsigned short* __restrict__ resid, // [V][1024][CPAD] bf16 (MODE 1)
    unsigned short* __restrict__ outb,        // bf16 out (MODE 0/1)
    float* __restrict__ outf)                 // fp32 out (MODE 2)
{
    // 8 output rows per block + 1-row halo each side: 10 rows x 32 x 64ch = 40 KB
    __shared__ __align__(16) unsigned short sIn[10 * 32 * CPAD];

    const int blk = blockIdx.x;
    const int v = blk >> 2;
    const int r0 = (blk & 3) * 8;               // first output row of this quarter

    // hint: weights are tiny and hot -> prefetch (global_prefetch_b8)
    __builtin_prefetch(wpk, 0, 3);

    const unsigned short* src = xin + (size_t)v * (HW * CPAD);
    // cooperative slab staging, 16B chunks: rows r0-1 .. r0+8 (zero-fill OOB rows)
#if USE_ASYNC_LDS
    {
        for (int ch = threadIdx.x; ch < (10 * 32 * CPAD) / 8; ch += 128) {
            const int elem = ch * 8;
            const int row = elem >> 11;             // / (32*64)
            const int y = r0 - 1 + row;
            if (y >= 0 && y < 32) {
                __builtin_amdgcn_global_load_async_to_lds_b128(
                    (gv4i*)(src + (size_t)y * (32 * CPAD) + (elem & (32 * CPAD - 1))),
                    (lv4i*)&sIn[elem], 0, 0);
            } else {
                *(v8us*)(&sIn[elem]) = (v8us){0, 0, 0, 0, 0, 0, 0, 0};
            }
        }
        __builtin_amdgcn_s_wait_asynccnt(0);
    }
#else
    for (int ch = threadIdx.x; ch < (10 * 32 * CPAD) / 8; ch += 128) {
        const int elem = ch * 8;
        const int row = elem >> 11;                 // / (32*64)
        const int y = r0 - 1 + row;
        v8us val = {0, 0, 0, 0, 0, 0, 0, 0};
        if (y >= 0 && y < 32)
            val = *(const v8us*)(src + (size_t)y * (32 * CPAD) + (elem & (32 * CPAD - 1)));
        *(v8us*)(&sIn[elem]) = val;
    }
#endif
    __syncthreads();

    const int lane = threadIdx.x & 31;
    const int wave = threadIdx.x >> 5;
    const int g = lane >> 4;                    // K-group (A/B lane-group per ISA layout)
    const int m = lane & 15;                    // M for A, N for B/D
    const int P0 = r0 * 32;

    for (int f = wave; f < 16 * NTILES; f += 4) {
        const int tile = f / NTILES;
        const int nt = f - tile * NTILES;
        const int P = P0 + tile * 16;           // 16 consecutive positions (half a row)
        const int p = P + m;
        const int yl = p >> 5;
        const int xp = p & 31;

        v8f acc = {0.f, 0.f, 0.f, 0.f, 0.f, 0.f, 0.f, 0.f};
        const unsigned short* wlane = wpk + (size_t)(nt * 16 + m) * CPAD;

        #pragma unroll
        for (int tap = 0; tap < 9; ++tap) {
            const int ky = tap / 3;
            const int kx = tap - ky * 3;
            const int ys = yl + ky - 1;         // always within slab rows
            const int xs = xp + kx - 1;
            const int lr = ys - (r0 - 1);       // 0..9
            const bool inb = (xs >= 0) && (xs < 32);
            const int lbase = (lr * 32 + xs) * CPAD;
            const unsigned short* wtap = wlane + (size_t)tap * OC * CPAD;
            #pragma unroll
            for (int c = 0; c < 2; ++c) {       // K chunks of 32 (ci 0..31, 32..63)
                BF16Frag a;
                if (inb) {
                    // ISA 16-bit A 16x32 layout: group g holds K = g*8+{0..7} and g*8+16+{0..7}
                    const int ci0 = c * 32 + g * 8;
                    a.h[0] = *(const v8us*)(&sIn[lbase + ci0]);
                    a.h[1] = *(const v8us*)(&sIn[lbase + ci0 + 16]);
                } else {
                    a.h[0] = (v8us){0, 0, 0, 0, 0, 0, 0, 0};
                    a.h[1] = (v8us){0, 0, 0, 0, 0, 0, 0, 0};
                }
                // B 32x16: lane group g holds K = g*8+{0..7} and g*8+16+{0..7} for N = lane&15
                BF16Frag b;
                const unsigned short* wb = wtap + c * 32 + g * 8;
                b.h[0] = *(const v8us*)(wb);
                b.h[1] = *(const v8us*)(wb + 16);
                acc = __builtin_amdgcn_wmma_f32_16x16x32_bf16(
                    false, a.v, false, b.v, (short)0, acc, false, false);
            }
        }

        // epilogue using C/D layout: VGPR r -> M = r + g*8, N = lane&15
        const int co = nt * 16 + m;
        const float bv = bias[co];
        #pragma unroll
        for (int r = 0; r < 8; ++r) {
            const int pp = P + r + g * 8;
            float val = acc[r] + bv;
            if (MODE == 0) {
                outb[(size_t)v * (HW * CPAD) + (size_t)pp * CPAD + co] = f2bf(fmaxf(val, 0.f));
            } else if (MODE == 1) {
                const size_t idx = (size_t)v * (HW * CPAD) + (size_t)pp * CPAD + co;
                val += bf2f(resid[idx]);
                outb[idx] = f2bf(val);
            } else {
                outf[((size_t)v * OC + co) * HW + pp] = val;
            }
        }
    }
}

// ---------------- 5. instance norm + relu ----------------
__global__ __launch_bounds__(256) void instnorm_relu(
    const float* __restrict__ yin, float* __restrict__ out)
{
    __shared__ float s1[256];
    __shared__ float s2[256];
    const int vc = blockIdx.x;                  // v*16 + c
    const float* p = yin + (size_t)vc * HW;
    const int t = threadIdx.x;
    const float a0 = p[t], a1 = p[t + 256], a2 = p[t + 512], a3 = p[t + 768];
    s1[t] = a0 + a1 + a2 + a3;
    s2[t] = a0 * a0 + a1 * a1 + a2 * a2 + a3 * a3;
    __syncthreads();
    for (int o = 128; o > 0; o >>= 1) {
        if (t < o) { s1[t] += s1[t + o]; s2[t] += s2[t + o]; }
        __syncthreads();
    }
    const float mean = s1[0] * (1.f / HW);
    const float var = s2[0] * (1.f / HW) - mean * mean;
    const float inv = rsqrtf(var + 1e-5f);
    float* o = out + (size_t)vc * HW;
    o[t]       = fmaxf((a0 - mean) * inv, 0.f);
    o[t + 256] = fmaxf((a1 - mean) * inv, 0.f);
    o[t + 512] = fmaxf((a2 - mean) * inv, 0.f);
    o[t + 768] = fmaxf((a3 - mean) * inv, 0.f);
}

// ---------------- host launch ----------------
extern "C" void kernel_launch(void* const* d_in, const int* in_sizes, int n_in,
                              void* d_out, int out_size, void* d_ws, size_t ws_size,
                              hipStream_t stream) {
    const float* feats = (const float*)d_in[0];
    const int*   edges = (const int*)d_in[1];
    const float* w1a = (const float*)d_in[2];  const float* b1a = (const float*)d_in[3];
    const float* w1b = (const float*)d_in[4];  const float* b1b = (const float*)d_in[5];
    const float* w2a = (const float*)d_in[6];  const float* b2a = (const float*)d_in[7];
    const float* w2b = (const float*)d_in[8];  const float* b2b = (const float*)d_in[9];
    const float* wf  = (const float*)d_in[10]; const float* bfp = (const float*)d_in[11];

    char* ws = (char*)d_ws;
    const size_t SZ = (size_t)NV * HW * CPAD * 2;      // 128 MB: one bf16 x-buffer
    // Region A: pooled_pos+pooled_neg (2x64MB fp32), later h (bf16)
    float* ppos = (float*)ws;
    float* pneg = (float*)(ws + SZ / 2);
    unsigned short* hbuf = (unsigned short*)ws;
    // Region B: x0, later x2
    unsigned short* x0 = (unsigned short*)(ws + SZ);
    // Region C: x1, later yws (fp32, 64MB)
    unsigned short* x1 = (unsigned short*)(ws + 2 * SZ);
    float* yws = (float*)(ws + 2 * SZ);
    // Weight packs past the big regions
    const size_t WSTEP = (size_t)9 * CIN * CPAD;       // elements per 48-ch layer
    unsigned short* wpk1a = (unsigned short*)(ws + 3 * SZ);
    unsigned short* wpk1b = wpk1a + WSTEP;
    unsigned short* wpk2a = wpk1b + WSTEP;
    unsigned short* wpk2b = wpk2a + WSTEP;
    unsigned short* wpkF  = wpk2b + WSTEP;

    // 1. pooled = 0, scatter
    (void)hipMemsetAsync(ws, 0, SZ, stream);
    pool_edges<<<dim3(4096, 64), 256, 0, stream>>>(feats, edges, ppos, pneg);

    // 2. x0 (zero-pad channels 48..63), pack
    (void)hipMemsetAsync(ws + SZ, 0, SZ, stream);
    pack_x<<<(int)(((size_t)NV * HW * CIN + 255) / 256), 256, 0, stream>>>(feats, ppos, pneg, x0);

    // 3. weight packs
    const int WG = (9 * CIN * CPAD + 255) / 256;
    pack_w<<<WG, 256, 0, stream>>>(w1a, wpk1a, CIN, CIN);
    pack_w<<<WG, 256, 0, stream>>>(w1b, wpk1b, CIN, CIN);
    pack_w<<<WG, 256, 0, stream>>>(w2a, wpk2a, CIN, CIN);
    pack_w<<<WG, 256, 0, stream>>>(w2b, wpk2b, CIN, CIN);
    pack_w<<<(9 * NC * CPAD + 255) / 256, 256, 0, stream>>>(wf, wpkF, NC, CIN);

    // 4. zero h and x1 regions (channel pads must read as 0)
    (void)hipMemsetAsync(ws, 0, SZ, stream);
    (void)hipMemsetAsync(ws + 2 * SZ, 0, SZ, stream);

    const int CB = NV * 4;   // one block per image-quarter
    // res block 1
    conv3x3_wmma<0, 3, CIN><<<CB, 128, 0, stream>>>(x0,   wpk1a, b1a, nullptr, hbuf, nullptr);
    conv3x3_wmma<1, 3, CIN><<<CB, 128, 0, stream>>>(hbuf, wpk1b, b1b, x0,      x1,   nullptr);
    // res block 2 (x2 overwrites region B)
    conv3x3_wmma<0, 3, CIN><<<CB, 128, 0, stream>>>(x1,   wpk2a, b2a, nullptr, hbuf, nullptr);
    conv3x3_wmma<1, 3, CIN><<<CB, 128, 0, stream>>>(hbuf, wpk2b, b2b, x1,      x0,   nullptr);
    // final conv 48->16 (fp32 NCHW into region C)
    conv3x3_wmma<2, 1, NC><<<CB, 128, 0, stream>>>(x0, wpkF, bfp, nullptr, nullptr, yws);

    // 5. instance norm + relu
    instnorm_relu<<<NV * NC, 256, 0, stream>>>(yws, (float*)d_out);
}
